// SmoothCRF_46815143526663
// MI455X (gfx1250) — matrix operand order repaired and checked by reading, not compiled, not run on hardware
//
#include <hip/hip_runtime.h>
#include <hip/hip_bf16.h>
#include <cstdint>

#define NT 96            // NUM_TAGS
#define BB 128           // batch
#define LL 2048          // seq len

typedef __attribute__((ext_vector_type(16))) __bf16 v16bf;
typedef __attribute__((ext_vector_type(8)))  float  v8f;

union AV16 { v16bf v; float4 f[2]; };

#if __has_builtin(__builtin_amdgcn_tensor_load_to_lds)
#define HAVE_TDM 1
typedef __attribute__((ext_vector_type(4))) unsigned int u32x4;
typedef __attribute__((ext_vector_type(8))) int          i32x8;
typedef __attribute__((ext_vector_type(4))) int          i32x4;
#endif

__device__ __forceinline__ unsigned short f2bf(float f) {
    unsigned int u = __float_as_uint(f);
    u += 0x7FFFu + ((u >> 16) & 1u);          // round-to-nearest-even
    return (unsigned short)(u >> 16);
}
__device__ __forceinline__ float bf2f(unsigned short u) {
    return __uint_as_float(((unsigned int)u) << 16);
}
__device__ __forceinline__ void atomicMaxF(float* a, float v) {
    if (v >= 0.f) atomicMax((int*)a, __float_as_int(v));
    else          atomicMin((unsigned int*)a, __float_as_uint(v));
}

#ifdef HAVE_TDM
// One TDM descriptor: 2-D tile, 16 rows (batch) x 96 f32 (tags),
// row stride = LL*NT elements.  D# packing per CDNA5 ISA §8.3/8.4.
// This toolchain's builtin is the 6-arg form:
//   (u32x4 group0, i32x8 group1, i32x4 group2, i32x4 group3, i32x8 extra, i32 cpol)
__device__ __forceinline__ void tdm_load_tile(const float* gp, unsigned lds_off) {
    unsigned long long ga = (unsigned long long)(uintptr_t)gp;
    u32x4 g0;
    g0[0] = 1u;                                     // count=1 (valid), user mode
    g0[1] = lds_off;                                // lds_addr (bytes)
    g0[2] = (unsigned)(ga & 0xFFFFFFFFu);           // global_addr[31:0]
    g0[3] = (unsigned)((ga >> 32) & 0x1FFFFFFu)     // global_addr[56:32]
          | (2u << 30);                             // type=2 ("image")
    i32x8 g1;
    g1[0] = 0x20000;                                // data_size=2 -> 4 bytes
    g1[1] = (int)(96u << 16);                       // tensor_dim0 = 96 (lo16 @bit48)
    g1[2] = (int)(16u << 16);                       // tensor_dim1 = 16 (lo16 @bit80)
    g1[3] = (int)(96u << 16);                       // tile_dim0   = 96 (@bit112)
    g1[4] = 16;                                     // tile_dim1   = 16 (@bit128)
    g1[5] = LL * NT;                                // tensor_dim0_stride (lo32 @bit160)
    g1[6] = 0;
    g1[7] = 0;
    i32x4 z4 = {0, 0, 0, 0};
    i32x8 z8 = {0, 0, 0, 0, 0, 0, 0, 0};
    __builtin_amdgcn_tensor_load_to_lds(g0, g1, z4, z4, z8, 0);
}
#endif

// ---------------------------------------------------------------- init
__global__ void init_kernel(float* scalars) {
    if (threadIdx.x < 2) scalars[threadIdx.x] = 0.f;
}

// exp(transitions) -> bf16, kept for the whole recurrence (B-matrix of the WMMA)
__global__ void expT_kernel(const float* __restrict__ trans,
                            unsigned short* __restrict__ expT) {
    int i = blockIdx.x * blockDim.x + threadIdx.x;
    if (i < NT * NT) expT[i] = f2bf(__expf(trans[i]));
}

// S[i,t] = sum_b emissions[b,i,t]   (one full 100MB streaming pass)
__global__ void sumB_kernel(const float* __restrict__ em, float* __restrict__ S) {
    int idx = blockIdx.x * blockDim.x + threadIdx.x;   // 0 .. L*NT-1
    if (idx >= LL * NT) return;
    float acc = 0.f;
    const float* p = em + idx;
    #pragma unroll 4
    for (int b = 0; b < BB; ++b) acc += p[(size_t)b * LL * NT];
    S[idx] = acc;
}

// per-sequence: length, E contribution (via S), transition path, start/end
__global__ void score_kernel(const float* __restrict__ S,
                             const float* __restrict__ trans,
                             const float* __restrict__ startT,
                             const float* __restrict__ endT,
                             const int* __restrict__ tags,
                             const unsigned char* __restrict__ mask,
                             float* __restrict__ scalars,
                             int* __restrict__ lengths) {
    const int c = blockIdx.x;
    const int tid = threadIdx.x;
    __shared__ float redf[256];
    __shared__ int   redi[256];

    int lacc = 0;
    for (int i = tid; i < LL; i += 256) lacc += mask[(size_t)c * LL + i] ? 1 : 0;
    redi[tid] = lacc; __syncthreads();
    for (int s = 128; s > 0; s >>= 1) { if (tid < s) redi[tid] += redi[tid + s]; __syncthreads(); }
    __shared__ int s_len;
    if (tid == 0) { s_len = redi[0]; lengths[c] = redi[0]; }
    __syncthreads();

    float eacc = 0.f;
    for (int i = tid; i < LL - 1; i += 256)
        if (mask[(size_t)c * LL + i]) eacc += S[i * NT + tags[(size_t)c * LL + i]];
    redf[tid] = eacc; __syncthreads();
    for (int s = 128; s > 0; s >>= 1) { if (tid < s) redf[tid] += redf[tid + s]; __syncthreads(); }
    float e_tot = redf[0]; __syncthreads();

    float tracc = 0.f;
    for (int i = tid + 1; i < LL; i += 256)
        if (mask[(size_t)c * LL + i])
            tracc += trans[tags[(size_t)c * LL + i - 1] * NT + tags[(size_t)c * LL + i]];
    redf[tid] = tracc; __syncthreads();
    for (int s = 128; s > 0; s >>= 1) { if (tid < s) redf[tid] += redf[tid + s]; __syncthreads(); }

    if (tid == 0) {
        atomicAdd(&scalars[0], e_tot);
        int len = s_len;
        float path = startT[tags[(size_t)c * LL]] + redf[0]
                   + endT[tags[(size_t)c * LL + len - 1]];
        atomicAdd(&scalars[1], path);
    }
}

// ---------------------------------------------------------------- forward recurrence
// alpha kept in scaled linear space (bf16 in LDS) + per-row f32 log-scale m.
// Per step: alpha_lin x exp(T)  -> WMMA bf16 GEMM (16x96 per block, 6 waves).
// Emission tiles for step t+1 are DMA'd by the Tensor Data Mover into a double
// buffer in LDS while step t computes (s_wait_tensorcnt before the barrier).
__global__ __launch_bounds__(192)
void forward_kernel(const float* __restrict__ em,
                    const float* __restrict__ startT,
                    const float* __restrict__ endT,
                    const unsigned short* __restrict__ expT,
                    const int* __restrict__ lengths,
                    float* __restrict__ partition) {
    __shared__ __align__(16) unsigned short s_alpha[16][104];  // 16 x 96 bf16 (+pad)
    __shared__ float s_rowmax[16];
    __shared__ float s_m[16];
    __shared__ float s_sum[16];
    __shared__ int   s_len[16];
#ifdef HAVE_TDM
    __shared__ __align__(16) float s_emit[2][16][NT];          // double-buffered tiles
#endif

    const int tid  = threadIdx.x;
    const int lane = tid & 31;
    const int wv   = tid >> 5;          // 0..5 : N tile (16 columns each)
    const int half = lane >> 4;         // 0/1
    const int lnl  = lane & 15;
    const int b0   = blockIdx.x * 16;   // this block's 16 batch rows
    const float* emb = em + (size_t)b0 * LL * NT;

    if (tid < 16) {
        s_len[tid]    = lengths[b0 + tid];
        s_rowmax[tid] = __uint_as_float(0xFF800000u);   // -inf
        s_sum[tid]    = 0.f;
    }
    __syncthreads();

#ifdef HAVE_TDM
    // kick the DMA for step t=1 immediately; it overlaps the t=0 setup below
    if (wv == 0)
        tdm_load_tile(emb + (size_t)1 * NT,
                      (unsigned)(uintptr_t)&s_emit[1][0][0]);
#endif

    // ---- t=0: alpha0 = start + emissions[:,0,:]; rescale to linear ----
    float v0[8];
    #pragma unroll
    for (int e = 0; e < 8; ++e) {
        int idx = tid + e * 192;                 // 16*96 = 1536 elements exactly
        int row = idx / NT, j = idx % NT;
        float val = startT[j] + emb[(size_t)row * LL * NT + j];
        v0[e] = val;
        atomicMaxF(&s_rowmax[row], val);
    }
    __syncthreads();
    #pragma unroll
    for (int e = 0; e < 8; ++e) {
        int idx = tid + e * 192;
        int row = idx / NT, j = idx % NT;
        s_alpha[row][j] = f2bf(__expf(v0[e] - s_rowmax[row]));
    }
    if (tid < 16) s_m[tid] = s_rowmax[tid];
    __syncthreads();
    if (tid < 16) s_rowmax[tid] = 0.f;

    // ---- preload this wave's exp(T) B-tiles (32x16 bf16 per K-chunk) ----
    v16bf bfrag[3];
    #pragma unroll
    for (int kc = 0; kc < 3; ++kc) {
        AV16 tmp;
        unsigned short u[16];
        #pragma unroll
        for (int h = 0; h < 16; ++h) {
            int K = kc * 32 + (h < 8 ? half * 8 + h : 16 + half * 8 + (h - 8));
            u[h] = expT[K * NT + wv * 16 + lnl];
        }
        tmp.f[0] = *(const float4*)&u[0];
        tmp.f[1] = *(const float4*)&u[8];
        bfrag[kc] = tmp.v;
    }
#ifdef HAVE_TDM
    if (wv == 0) __builtin_amdgcn_s_wait_tensorcnt(0);  // t=1 tile resident
#endif
    __syncthreads();

    // ---- sequential scan over time ----
    for (int t = 1; t < LL; ++t) {
        const int jj = wv * 16 + lnl;
#ifdef HAVE_TDM
        const int buf = t & 1;
        // issue DMA for t+1 into the other buffer (consumed in the previous step)
        if (wv == 0 && t + 1 < LL)
            tdm_load_tile(emb + (size_t)(t + 1) * NT,
                          (unsigned)(uintptr_t)&s_emit[buf ^ 1][0][0]);
#endif
        // A-frags from LDS, accumulate 3 K-chunks into f32 C
        v8f c = {};
        #pragma unroll
        for (int kc = 0; kc < 3; ++kc) {
            AV16 a;
            const unsigned short* rp = s_alpha[lnl];
            a.f[0] = *(const float4*)(rp + kc * 32 + half * 8);
            a.f[1] = *(const float4*)(rp + kc * 32 + 16 + half * 8);
            c = __builtin_amdgcn_wmma_f32_16x16x32_bf16(
                    false, a.v, false, bfrag[kc], (short)0, c, false, false);
        }

        // apply emissions in linear space; C layout: VGPR r -> M = half*8+r, N = lnl
        float vals[8], lmax[8];
        #pragma unroll
        for (int r = 0; r < 8; ++r) {
            int M = half * 8 + r;
#ifdef HAVE_TDM
            float emit = s_emit[buf][M][jj];
#else
            size_t eix = (size_t)M * LL * NT + (size_t)t * NT + jj;
            float emit = emb[eix];
            if (t + 1 < LL) __builtin_prefetch(&emb[eix + NT], 0, 1);
#endif
            vals[r] = c[r] * __expf(emit);        // >= 0
            lmax[r] = vals[r];
        }
        // row-max across the 16 lanes holding each row
        #pragma unroll
        for (int off = 1; off < 16; off <<= 1) {
            #pragma unroll
            for (int r = 0; r < 8; ++r)
                lmax[r] = fmaxf(lmax[r], __shfl_xor(lmax[r], off, 16));
        }
        if (lnl == 0) {
            #pragma unroll
            for (int r = 0; r < 8; ++r)
                atomicMax((int*)&s_rowmax[half * 8 + r], __float_as_int(lmax[r]));
        }
        __syncthreads();   // rowmax done; all A-frag reads done -> safe to overwrite alpha

        #pragma unroll
        for (int r = 0; r < 8; ++r) {
            int M = half * 8 + r;
            if (t < s_len[M]) {                   // masked rows keep old alpha & m
                float rm = fmaxf(s_rowmax[M], 1e-30f);
                s_alpha[M][jj] = f2bf(vals[r] / rm);
            }
        }
        if (tid < 16 && t < s_len[tid])
            s_m[tid] += __logf(fmaxf(s_rowmax[tid], 1e-30f));
#ifdef HAVE_TDM
        if (wv == 0) __builtin_amdgcn_s_wait_tensorcnt(0);  // t+1 tile resident
#endif
        __syncthreads();   // alpha + next emission tile visible to all waves
        if (tid < 16) s_rowmax[tid] = 0.f;
        __syncthreads();
    }

    // ---- partition[b] = m[b] + log( sum_j alpha_lin[b,j] * exp(end[j]) ) ----
    if (tid < NT) {
        float e = __expf(endT[tid]);
        #pragma unroll 4
        for (int r = 0; r < 16; ++r)
            atomicAdd(&s_sum[r], bf2f(s_alpha[r][tid]) * e);
    }
    __syncthreads();
    if (tid < 16)
        partition[b0 + tid] = s_m[tid] + __logf(fmaxf(s_sum[tid], 1e-37f));
}

// ---------------------------------------------------------------- final scalar
__global__ void final_kernel(const float* __restrict__ partition,
                             const float* __restrict__ scalars,
                             float* __restrict__ out) {
    __shared__ float red[128];
    int tid = threadIdx.x;
    red[tid] = partition[tid];
    __syncthreads();
    for (int s = 64; s > 0; s >>= 1) { if (tid < s) red[tid] += red[tid + s]; __syncthreads(); }
    if (tid == 0) {
        float meanE    = scalars[0] / (float)(BB * BB);
        float meanPath = scalars[1] / (float)BB;
        float meanPart = red[0]     / (float)BB;
        out[0] = meanE + meanPath - meanPart;
    }
}

// ---------------------------------------------------------------- launch
extern "C" void kernel_launch(void* const* d_in, const int* in_sizes, int n_in,
                              void* d_out, int out_size, void* d_ws, size_t ws_size,
                              hipStream_t stream) {
    const float*         em     = (const float*)d_in[0];
    const float*         trans  = (const float*)d_in[1];
    const float*         startT = (const float*)d_in[2];
    const float*         endT   = (const float*)d_in[3];
    const int*           tags   = (const int*)d_in[4];
    const unsigned char* mask   = (const unsigned char*)d_in[5];
    float*               out    = (float*)d_out;

    float* ws_f      = (float*)d_ws;
    float* S         = ws_f;                         // L*NT = 196608 floats
    float* partition = ws_f + 196608;                // 128
    float* scalars   = ws_f + 196736;                // 2 (E_sum, path_sum)
    int*   lengths   = (int*)(ws_f + 196740);        // 128
    unsigned short* expT = (unsigned short*)(ws_f + 196880);  // 96*96 bf16

    init_kernel <<<1, 64, 0, stream>>>(scalars);
    expT_kernel <<<36, 256, 0, stream>>>(trans, expT);
    sumB_kernel <<<768, 256, 0, stream>>>(em, S);
    score_kernel<<<128, 256, 0, stream>>>(S, trans, startT, endT, tags, mask,
                                          scalars, lengths);
    forward_kernel<<<8, 192, 0, stream>>>(em, startT, endT, expT, lengths, partition);
    final_kernel<<<1, 128, 0, stream>>>(partition, scalars, out);
}
